// TemplatePointwiseAttention_14542759264355
// MI455X (gfx1250) — matrix operand order repaired and checked by reading, not compiled, not run on hardware
//
#include <hip/hip_runtime.h>
#include <cstdint>

typedef __bf16 v16bf __attribute__((ext_vector_type(16)));
typedef __bf16 v2bf  __attribute__((ext_vector_type(2)));
typedef float  v8f   __attribute__((ext_vector_type(8)));

#define L_DIM 384
#define PAIRS (L_DIM*L_DIM)   // 147456
#define CZ 128
#define CT 64
#define T_N 4

// ws bf16 fragment buffer offsets (elements). Layout per region: [nt][kk][lane(32)][slot(16)]
#define WQF_OFF 0       // Wq: K=128 N=64  -> 4 ntiles x 4 ktiles  = 8192 elems
#define WKF_OFF 8192    // Wk: K=64  N=64  -> 4 x 2               = 4096
#define WVF_OFF 12288   // Wv: same as Wk                          = 4096
#define WOF_OFF 16384   // Wo: K=64  N=128 -> 8 x 2               = 8192
#define WSF_TOTAL 24576

// ---------- bf16 conversion helpers ----------
__device__ __forceinline__ uint16_t f2bf_manual(float x){
  uint32_t u = __float_as_uint(x);
  uint32_t r = u + 0x7FFFu + ((u >> 16) & 1u);   // round-to-nearest-even
  return (uint16_t)(r >> 16);
}
// pack two floats -> two bf16 in one u32 (v_cvt_pk_bf16_f32 when available)
__device__ __forceinline__ uint32_t f2bf2(float lo, float hi){
#if __has_builtin(__builtin_amdgcn_cvt_pk_bf16_f32)
  union { v2bf v; uint32_t u; } cv;
  cv.v = __builtin_amdgcn_cvt_pk_bf16_f32(lo, hi);
  return cv.u;
#else
  return (uint32_t)f2bf_manual(lo) | ((uint32_t)f2bf_manual(hi) << 16);
#endif
}
__device__ __forceinline__ uint16_t f2bf(float x){
#if __has_builtin(__builtin_amdgcn_cvt_pk_bf16_f32)
  return (uint16_t)(f2bf2(x, x) & 0xFFFFu);
#else
  return f2bf_manual(x);
#endif
}
__device__ __forceinline__ float bf2f(uint16_t b){
  return __uint_as_float(((uint32_t)b) << 16);
}

union Frag16 {
  v16bf bf;
  uint32_t u[8];
  uint4 q[2];
};

__device__ __forceinline__ v8f vzero(){
  v8f z;
  #pragma unroll
  for (int i = 0; i < 8; ++i) z[i] = 0.f;
  return z;
}

// A-matrix K striping within a 32-wide chunk (16-bit A 16x32, ISA 7.12.2):
// VGPR0-3 = K[h*8 .. h*8+7], VGPR4-7 = K[16+h*8 .. 16+h*8+7]
__device__ __forceinline__ int klocA(int slot, int h){
  return (slot < 8) ? (h*8 + slot) : (16 + h*8 + (slot - 8));
}
// Contiguous K order (not AMD's table order; unused, B is packed with klocA):
// lanes 0-15 hold K=0..15, lanes 16-31 hold K=16..31, two per VGPR -> k = h*16 + slot
__device__ __forceinline__ int klocB(int slot, int h){
  return h*16 + slot;
}

// ---------- setup: swizzle fp32 weights into WMMA-B bf16 fragments in d_ws ----------
__global__ __launch_bounds__(256) void wswz(const float* __restrict__ Wq,
                                            const float* __restrict__ Wk,
                                            const float* __restrict__ Wv,
                                            const float* __restrict__ Wo,
                                            uint16_t* __restrict__ ws){
  int id = blockIdx.x * blockDim.x + threadIdx.x;
  if (id >= WSF_TOTAL) return;
  const float* W; int N, ktiles, base;
  if (id < WKF_OFF)      { W = Wq; N = 64;  ktiles = 4; base = WQF_OFF; }
  else if (id < WVF_OFF) { W = Wk; N = 64;  ktiles = 2; base = WKF_OFF; }
  else if (id < WOF_OFF) { W = Wv; N = 64;  ktiles = 2; base = WVF_OFF; }
  else                   { W = Wo; N = 128; ktiles = 2; base = WOF_OFF; }
  int lid  = id - base;
  int slot = lid & 15;
  int lane = (lid >> 4) & 31;
  int fk   = lid >> 9;
  int kk   = fk % ktiles;
  int nt   = fk / ktiles;
  int k = kk*32 + klocA(slot, lane >> 4);
  int n = nt*16 + (lane & 15);
  ws[id] = f2bf(W[k*N + n]);
}

// ---------- fragment loads ----------
__device__ __forceinline__ Frag16 loadB(const uint16_t* __restrict__ ws,
                                        int base, int ktiles, int nt, int kk, int lane){
  Frag16 f;
  const uint4* p = (const uint4*)(ws + base + (((nt*ktiles + kk)*32 + lane) << 4));
  f.q[0] = p[0];
  f.q[1] = p[1];
  return f;
}

__device__ __forceinline__ Frag16 loadA(const uint16_t* act, int Kd, int m, int h, int kk){
  Frag16 f;
  const uint16_t* row = act + m*Kd + kk*32;
  #pragma unroll
  for (int s = 0; s < 4; ++s)
    f.u[s] = *(const uint32_t*)(row + h*8 + 2*s);
  #pragma unroll
  for (int s = 0; s < 4; ++s)
    f.u[4+s] = *(const uint32_t*)(row + 16 + h*8 + 2*s);
  return f;
}

#define WMMA_BF16(A, B, C) \
  __builtin_amdgcn_wmma_f32_16x16x32_bf16(false, (A), false, (B), (short)0, (C), false, false)

// ---------- fused main kernel: 1 wave = 16 pairs ----------
__global__ __launch_bounds__(128) void tpa_main(
    const float* __restrict__ tin, const float* __restrict__ zin,
    const float* __restrict__ tmask,
    const float* __restrict__ zg, const float* __restrict__ zb,
    const float* __restrict__ tg, const float* __restrict__ tb,
    const float* __restrict__ bo,
    const uint16_t* __restrict__ ws,
    float* __restrict__ out)
{
  __shared__ __align__(16) uint16_t s_zn[4][16*CZ];       // bf16 normalized z rows
  __shared__ __align__(16) uint16_t s_tn[4][16*CT];       // bf16 normalized t rows / o buffer
  __shared__ __align__(16) uint16_t s_vb[4][T_N*16*CT];   // bf16 v[t][p][hc]
  __shared__ float s_lgt[4][16*T_N*4];                    // logits -> attn [p][t][h]

  const int lane = threadIdx.x & 31;
  const int wv   = threadIdx.x >> 5;
  const int h    = lane >> 4;        // lane half
  const int col  = lane & 15;
  const int pt   = (blockIdx.x*4 + wv)*16;   // first pair of this wave's tile

  uint16_t* zn = &s_zn[wv][0];
  uint16_t* tn = &s_tn[wv][0];
  uint16_t* vb = &s_vb[wv][0];
  float*    lg = &s_lgt[wv][0];

  // ---------------- LayerNorm(z) -> LDS bf16 ----------------
  for (int r = 0; r < 16; ++r) {
    const float* row = zin + (size_t)(pt + r)*CZ;
    float4 xv = *(const float4*)(row + lane*4);
    // prefetch first template's rows while doing z work
    __builtin_prefetch(tin + (size_t)(pt + r)*CT + lane*2, 0, 1);
    float s1 = xv.x + xv.y + xv.z + xv.w;
    float s2 = xv.x*xv.x + xv.y*xv.y + xv.z*xv.z + xv.w*xv.w;
    #pragma unroll
    for (int m = 1; m < 32; m <<= 1) {
      s1 += __shfl_xor(s1, m, 32);
      s2 += __shfl_xor(s2, m, 32);
    }
    float mean = s1 * (1.f/128.f);
    float var  = s2 * (1.f/128.f) - mean*mean;
    float rinv = rsqrtf(var + 1e-5f);
    float4 gv = *(const float4*)(zg + lane*4);
    float4 bv = *(const float4*)(zb + lane*4);
    uint32_t* d = (uint32_t*)(zn + r*CZ + lane*4);
    d[0] = f2bf2((xv.x - mean)*rinv*gv.x + bv.x, (xv.y - mean)*rinv*gv.y + bv.y);
    d[1] = f2bf2((xv.z - mean)*rinv*gv.z + bv.z, (xv.w - mean)*rinv*gv.w + bv.w);
  }
  __syncthreads();

  // ---------------- q = zn @ Wq  (16x64) : frags stay in VGPRs ----------------
  Frag16 az[4];
  #pragma unroll
  for (int kk = 0; kk < 4; ++kk) az[kk] = loadA(zn, CZ, col, h, kk);

  v8f qf[4];
  #pragma unroll
  for (int nt = 0; nt < 4; ++nt) {
    v8f acc = vzero();
    #pragma unroll
    for (int kk = 0; kk < 4; ++kk) {
      Frag16 b = loadB(ws, WQF_OFF, 4, nt, kk, lane);
      acc = WMMA_BF16(az[kk].bf, b.bf, acc);
    }
    qf[nt] = acc;
  }

  // ---------------- per-template: LN(t), k GEMM -> logits, v GEMM -> LDS ----------------
  for (int tt = 0; tt < T_N; ++tt) {
    for (int r = 0; r < 16; ++r) {
      const float* row = tin + ((size_t)tt*PAIRS + pt + r)*CT;
      float2 xv = *(const float2*)(row + lane*2);
      // prefetch next template's tile rows into cache while computing this one
      if (tt < T_N-1)
        __builtin_prefetch(tin + ((size_t)(tt+1)*PAIRS + pt + r)*CT + lane*2, 0, 1);
      float s1 = xv.x + xv.y;
      float s2 = xv.x*xv.x + xv.y*xv.y;
      #pragma unroll
      for (int m = 1; m < 32; m <<= 1) {
        s1 += __shfl_xor(s1, m, 32);
        s2 += __shfl_xor(s2, m, 32);
      }
      float mean = s1 * (1.f/64.f);
      float var  = s2 * (1.f/64.f) - mean*mean;
      float rinv = rsqrtf(var + 1e-5f);
      float2 gv = *(const float2*)(tg + lane*2);
      float2 bv = *(const float2*)(tb + lane*2);
      uint32_t* d = (uint32_t*)(tn + r*CT + lane*2);
      d[0] = f2bf2((xv.x - mean)*rinv*gv.x + bv.x, (xv.y - mean)*rinv*gv.y + bv.y);
    }
    __syncthreads();

    Frag16 at[2];
    at[0] = loadA(tn, CT, col, h, 0);
    at[1] = loadA(tn, CT, col, h, 1);

    #pragma unroll
    for (int nt = 0; nt < 4; ++nt) {      // nt == head (C=16 == N-tile width)
      // k tile + logits
      v8f kacc = vzero();
      #pragma unroll
      for (int kk = 0; kk < 2; ++kk) {
        Frag16 b = loadB(ws, WKF_OFF, 2, nt, kk, lane);
        kacc = WMMA_BF16(at[kk].bf, b.bf, kacc);
      }
      v8f pr = qf[nt] * kacc;             // elementwise q*k over this head's 16 channels
      #pragma unroll
      for (int m = 1; m < 16; m <<= 1) {  // reduce across the 16 lanes of each half
        #pragma unroll
        for (int r = 0; r < 8; ++r) pr[r] += __shfl_xor(pr[r], m, 32);
      }
      if (col == 0) {
        #pragma unroll
        for (int r = 0; r < 8; ++r)
          lg[((h*8 + r)*T_N + tt)*4 + nt] = pr[r] * 0.25f;   // /sqrt(C)
      }
      // v tile -> LDS bf16
      v8f vacc = vzero();
      #pragma unroll
      for (int kk = 0; kk < 2; ++kk) {
        Frag16 b = loadB(ws, WVF_OFF, 2, nt, kk, lane);
        vacc = WMMA_BF16(at[kk].bf, b.bf, vacc);
      }
      #pragma unroll
      for (int r = 0; r < 8; ++r)
        vb[(tt*16 + h*8 + r)*CT + nt*16 + col] = f2bf(vacc[r]);
    }
    __syncthreads();
  }

  // ---------------- masked softmax over T + renormalize ----------------
  #pragma unroll
  for (int it = 0; it < 2; ++it) {
    int lin = it*32 + lane;      // 64 (pair, head) combos
    int p  = lin >> 2;
    int hh = lin & 3;
    int gp = pt + p;
    int i  = gp / L_DIM;
    int j  = gp - i*L_DIM;
    float l[4], pm[4];
    #pragma unroll
    for (int t4 = 0; t4 < 4; ++t4) {
      pm[t4] = tmask[t4*L_DIM + i] * tmask[t4*L_DIM + j];
      float lv = lg[(p*T_N + t4)*4 + hh];
      l[t4] = (pm[t4] == 0.f) ? -1e9f : lv;
    }
    float mx = fmaxf(fmaxf(l[0], l[1]), fmaxf(l[2], l[3]));
    float e[4], s = 0.f;
    #pragma unroll
    for (int t4 = 0; t4 < 4; ++t4) { e[t4] = __expf(l[t4] - mx); s += e[t4]; }
    float inv = 1.f / s;
    float s2 = 0.f;
    #pragma unroll
    for (int t4 = 0; t4 < 4; ++t4) { e[t4] = e[t4]*inv*pm[t4]; s2 += e[t4]; }
    float r2 = 1.f / fmaxf(s2, 1e-8f);
    #pragma unroll
    for (int t4 = 0; t4 < 4; ++t4)
      lg[(p*T_N + t4)*4 + hh] = e[t4] * r2;
  }
  __syncthreads();

  // ---------------- o = sum_t attn * v  -> bf16 in tn buffer ----------------
  // each lane handles 2 adjacent channels -> packed u32 LDS traffic
  #pragma unroll
  for (int j2 = 0; j2 < 16; ++j2) {
    int lin = j2*64 + lane*2;    // 1024 outputs = 16 pairs x 64 channels
    int p  = lin >> 6;
    int n  = lin & 63;           // even
    int hh = n >> 4;
    float a0 = 0.f, a1 = 0.f;
    #pragma unroll
    for (int t4 = 0; t4 < 4; ++t4) {
      float w = lg[(p*T_N + t4)*4 + hh];
      uint32_t vp = *(const uint32_t*)(vb + (t4*16 + p)*CT + n);
      a0 += w * bf2f((uint16_t)(vp & 0xFFFFu));
      a1 += w * bf2f((uint16_t)(vp >> 16));
    }
    *(uint32_t*)(tn + p*CT + n) = f2bf2(a0, a1);
  }
  __syncthreads();

  // ---------------- out = o @ Wo + bo  (16x128, fp32 stores) ----------------
  Frag16 ao[2];
  ao[0] = loadA(tn, CT, col, h, 0);
  ao[1] = loadA(tn, CT, col, h, 1);
  #pragma unroll
  for (int nt = 0; nt < 8; ++nt) {
    v8f acc = vzero();
    #pragma unroll
    for (int kk = 0; kk < 2; ++kk) {
      Frag16 b = loadB(ws, WOF_OFF, 2, nt, kk, lane);
      acc = WMMA_BF16(ao[kk].bf, b.bf, acc);
    }
    int n = nt*16 + col;
    float bias = bo[n];
    #pragma unroll
    for (int r = 0; r < 8; ++r)
      out[(size_t)(pt + h*8 + r)*CZ + n] = acc[r] + bias;
  }
}

extern "C" void kernel_launch(void* const* d_in, const int* in_sizes, int n_in,
                              void* d_out, int out_size, void* d_ws, size_t ws_size,
                              hipStream_t stream) {
  const float* t  = (const float*)d_in[0];
  const float* z  = (const float*)d_in[1];
  const float* tm = (const float*)d_in[2];
  const float* zg = (const float*)d_in[3];
  const float* zb = (const float*)d_in[4];
  const float* tg = (const float*)d_in[5];
  const float* tb = (const float*)d_in[6];
  const float* Wq = (const float*)d_in[7];
  const float* Wk = (const float*)d_in[8];
  const float* Wv = (const float*)d_in[9];
  const float* Wo = (const float*)d_in[10];
  const float* bo = (const float*)d_in[11];
  uint16_t* ws = (uint16_t*)d_ws;
  float* out = (float*)d_out;

  wswz<<<(WSF_TOTAL + 255)/256, 256, 0, stream>>>(Wq, Wk, Wv, Wo, ws);
  // 147456 pairs / (4 waves * 16 pairs) = 2304 blocks of 128 threads
  tpa_main<<<PAIRS/64, 128, 0, stream>>>(t, z, tm, zg, zb, tg, tb, bo, ws, out);
}
